// HyperbolicGRU_43095701848672
// MI455X (gfx1250) — matrix operand order repaired and checked by reading, not compiled, not run on hardware
//
#include <hip/hip_runtime.h>
#include <hip/hip_bf16.h>
#include <math.h>

typedef __bf16 bf16_t;
typedef __attribute__((ext_vector_type(16))) __bf16 v16bf;
typedef __attribute__((ext_vector_type(8)))  __bf16 v8bf;
typedef __attribute__((ext_vector_type(8)))  float   v8f;

#define BATCH 256
#define SEQ   512
#define DIMI  1024
#define DIMH  1024
#define N3H   3072   // 3*H
#define LDG   6144   // gx(3072) | gh(3072) per row

// ---------------------------------------------------------------------------
// One-time: fp32 weights -> bf16, concatenated [W_ih ; W_hh], row-major (3H x 1024)
// ---------------------------------------------------------------------------
__global__ void hgru_convert_w(const float* __restrict__ wi,
                               const float* __restrict__ wh,
                               bf16_t* __restrict__ wc) {
  const int NW = N3H * DIMI;
  int idx = blockIdx.x * blockDim.x + threadIdx.x;
  int stride = gridDim.x * blockDim.x;
  for (int i = idx; i < 2 * NW; i += stride) {
    float v = (i < NW) ? wi[i] : wh[i - NW];
    wc[i] = (bf16_t)v;
  }
}

// Zero the h_tan carry buffers (log_0(h0=0) = 0)
__global__ void hgru_zero_h(float* __restrict__ hf, bf16_t* __restrict__ hb, int n) {
  int i = blockIdx.x * blockDim.x + threadIdx.x;
  if (i < n) { hf[i] = 0.0f; hb[i] = (bf16_t)0.0f; }
}

// Gather h_n = out[:, S-1, :] into the concatenated tail of d_out
__global__ void hgru_hn(const float* __restrict__ out, float* __restrict__ hn) {
  int i = blockIdx.x * blockDim.x + threadIdx.x;
  if (i < BATCH * DIMH) {
    int m = i >> 10, j = i & (DIMH - 1);
    hn[i] = out[(size_t)m * SEQ * DIMH + (size_t)(SEQ - 1) * DIMH + j];
  }
}

// ---------------------------------------------------------------------------
// Per-step x prep: log_map_zero on x[:,t,:] rows only (h_tan comes fused from
// the gates kernel). Block = 256 threads per row of 1024 floats.
// ---------------------------------------------------------------------------
__global__ void hgru_prep_x(const float* __restrict__ x,
                            bf16_t* __restrict__ xt,
                            int t) {
  const int row = blockIdx.x;            // 0..255
  const float* src = x + (size_t)row * SEQ * DIMI + (size_t)t * DIMI;
  const int tid = threadIdx.x;

  float4 v = ((const float4*)src)[tid];
  float ss = v.x * v.x + v.y * v.y + v.z * v.z + v.w * v.w;
  for (int m = 16; m >= 1; m >>= 1) ss += __shfl_xor(ss, m, 32);
  __shared__ float red[8];
  if ((tid & 31) == 0) red[tid >> 5] = ss;
  __syncthreads();
  float total = 0.0f;
#pragma unroll
  for (int w = 0; w < 8; ++w) total += red[w];

  const float sc = 0.1f;                           // sqrt(0.01)
  float nrm = fmaxf(sqrtf(total), 1e-10f);
  float arg = fminf(sc * nrm, 1.0f - 1e-5f);
  float at  = 0.5f * logf((1.0f + arg) / (1.0f - arg));   // atanh
  float scl = at / (sc * nrm);

  bf16_t* xb = xt + (size_t)row * DIMI + tid * 4;
  xb[0] = (bf16_t)(v.x * scl); xb[1] = (bf16_t)(v.y * scl);
  xb[2] = (bf16_t)(v.z * scl); xb[3] = (bf16_t)(v.w * scl);
}

// ---------------------------------------------------------------------------
// GEMM: g[:, z*3072 .. ] = A(256x1024) @ W^T(1024x3072) + bias
//   z==0: A=x_tan, W=W_ih, b=b_ih   z==1: A=h_tan, W=W_hh, b=b_hh
// Block = 256 threads = 8 waves; wave owns a 16(M) x 64(N) strip.
// grid = (3072/512, 256/16, 2) = (6, 16, 2)
// ---------------------------------------------------------------------------
__global__ void hgru_gemm(const bf16_t* __restrict__ xt,
                          const bf16_t* __restrict__ htb,
                          const bf16_t* __restrict__ Wc,
                          const float* __restrict__ b_ih,
                          const float* __restrict__ b_hh,
                          float* __restrict__ g) {
  const int z = blockIdx.z;
  const bf16_t* A    = z ? htb : xt;
  const bf16_t* W    = Wc + (size_t)z * N3H * DIMI;
  const float*  bias = z ? b_hh : b_ih;

  const int wave = threadIdx.x >> 5;
  const int lane = threadIdx.x & 31;
  const int hi   = lane >> 4;        // half-wave select
  const int lo   = lane & 15;

  const int tm  = blockIdx.y;                    // M tile (16 rows)
  const int tn0 = blockIdx.x * 32 + wave * 4;    // first of 4 N tiles
  const int m   = tm * 16 + lo;                  // A row owned by this lane

  v8f acc[4] = {};

  for (int kk = 0; kk < DIMI; kk += 32) {
    // A fragment (16x32 bf16): lane holds K = {hi*16..hi*16+15}
    const bf16_t* arow = A + (size_t)m * DIMI + kk + hi * 16;
    v8bf a0 = *(const v8bf*)(arow);
    v8bf a1 = *(const v8bf*)(arow + 8);
    v16bf a;
#pragma unroll
    for (int e = 0; e < 8; ++e) { a[e] = a0[e]; a[8 + e] = a1[e]; }

#pragma unroll
    for (int j = 0; j < 4; ++j) {
      // B fragment (32x16 bf16): lane owns column n, 16 contiguous K of W row n
      const int n = (tn0 + j) * 16 + lo;
      v16bf b = *(const v16bf*)(W + (size_t)n * DIMI + kk + hi * 16);
      acc[j] = __builtin_amdgcn_wmma_f32_16x16x32_bf16(
                   false, a, false, b, (short)0, acc[j], false, false);
    }
  }

  // C layout: VGPR r -> row r + 8*hi, col = lo
  const int rowbase = tm * 16 + hi * 8;
#pragma unroll
  for (int j = 0; j < 4; ++j) {
    const int n  = (tn0 + j) * 16 + lo;
    const float bv = bias[n];
#pragma unroll
    for (int r = 0; r < 8; ++r) {
      g[(size_t)(rowbase + r) * LDG + (size_t)z * N3H + n] = acc[j][r] + bv;
    }
  }
}

// ---------------------------------------------------------------------------
// Gates + exp_map_zero, FUSED with next-step log_map_zero of h_new.
//   v = h_new_tan;  arg = min(sc*||v||, 15);  scl = tanh(arg)/(sc*||v||)
//   h_new = v*scl  (written to out[:,t,:])
//   ||h_new|| = tanh(arg)/sc  =>  next h_tan = v * scl * atanh(min(tanh(arg),
//   1-1e-5)) / tanh(arg)   -- no second reduction needed.
// Block per row (256 threads x 4 elems).
// ---------------------------------------------------------------------------
__global__ void hgru_gates(const float* __restrict__ g,
                           const float* __restrict__ htf_in,
                           float* __restrict__ htf_out,
                           bf16_t* __restrict__ htb_out,
                           float* __restrict__ out,
                           int t) {
  const int row = blockIdx.x;
  const int tid = threadIdx.x;
  const float* gr = g + (size_t)row * LDG;
  const float* hf = htf_in + (size_t)row * DIMH;

  float vtan[4];
  float ss = 0.0f;
#pragma unroll
  for (int e = 0; e < 4; ++e) {
    const int j = tid * 4 + e;
    float xr = gr[j],            xz = gr[DIMH + j],     xn = gr[2 * DIMH + j];
    float hr = gr[3 * DIMH + j], hz = gr[4 * DIMH + j], hn = gr[5 * DIMH + j];
    float r  = 1.0f / (1.0f + __expf(-(xr + hr)));
    float zz = 1.0f / (1.0f + __expf(-(xz + hz)));
    float nn = tanhf(xn + r * hn);
    float v  = (1.0f - zz) * nn + zz * hf[j];
    vtan[e] = v;
    ss += v * v;
  }
  for (int m = 16; m >= 1; m >>= 1) ss += __shfl_xor(ss, m, 32);
  __shared__ float red[8];
  if ((tid & 31) == 0) red[tid >> 5] = ss;
  __syncthreads();
  float total = 0.0f;
#pragma unroll
  for (int w = 0; w < 8; ++w) total += red[w];

  const float sc = 0.1f;
  float nrm = fmaxf(sqrtf(total), 1e-10f);
  float arg = fminf(sc * nrm, 15.0f);
  float th  = tanhf(arg);
  float scl = th / (sc * nrm);                 // exp_map scale

  // fused log_map of h_new: norm2 = th/sc, arg2 = min(th, 1-1e-5)
  float arg2 = fminf(th, 1.0f - 1e-5f);
  float at2  = 0.5f * logf((1.0f + arg2) / (1.0f - arg2));  // atanh
  float scl2 = at2 / th;                       // log_map scale on h_new
  float sclh = scl * scl2;                     // direct v -> next h_tan

#pragma unroll
  for (int e = 0; e < 4; ++e) {
    const int j = tid * 4 + e;
    float o  = vtan[e] * scl;                  // hyperbolic h (output)
    float ht = vtan[e] * sclh;                 // next-step tangent h
    out[(size_t)row * SEQ * DIMH + (size_t)t * DIMH + j] = o;
    htf_out[(size_t)row * DIMH + j] = ht;
    htb_out[(size_t)row * DIMH + j] = (bf16_t)ht;
  }
}

// ---------------------------------------------------------------------------
extern "C" void kernel_launch(void* const* d_in, const int* in_sizes, int n_in,
                              void* d_out, int out_size, void* d_ws, size_t ws_size,
                              hipStream_t stream) {
  const float* x    = (const float*)d_in[0];  // (B,S,I)
  const float* W_ih = (const float*)d_in[1];  // (3H,I)
  const float* W_hh = (const float*)d_in[2];  // (3H,H)
  const float* b_ih = (const float*)d_in[3];  // (3H)
  const float* b_hh = (const float*)d_in[4];  // (3H)
  float* out = (float*)d_out;                 // (B,S,H) then (1,B,H)

  char* ws = (char*)d_ws;
  bf16_t* Wc  = (bf16_t*)(ws + 0);                 // 2*3072*1024*2 = 12,582,912 B
  bf16_t* xt  = (bf16_t*)(ws + 12582912);          // 256*1024*2    =    524,288 B
  bf16_t* htb = (bf16_t*)(ws + 13107200);          // 256*1024*2    =    524,288 B
  float*  htf = (float*) (ws + 13631488);          // 256*1024*4    =  1,048,576 B
  float*  g   = (float*) (ws + 14680064);          // 256*6144*4    =  6,291,456 B

  hgru_convert_w<<<dim3(6144), dim3(256), 0, stream>>>(W_ih, W_hh, Wc);
  hgru_zero_h<<<dim3((BATCH * DIMH + 255) / 256), dim3(256), 0, stream>>>(
      htf, htb, BATCH * DIMH);

  for (int t = 0; t < SEQ; ++t) {
    hgru_prep_x<<<dim3(BATCH), dim3(256), 0, stream>>>(x, xt, t);
    hgru_gemm<<<dim3(6, 16, 2), dim3(256), 0, stream>>>(xt, htb, Wc, b_ih, b_hh, g);
    hgru_gates<<<dim3(BATCH), dim3(256), 0, stream>>>(g, htf, htf, htb, out, t);
  }

  hgru_hn<<<dim3((BATCH * DIMH + 255) / 256), dim3(256), 0, stream>>>(
      out, out + (size_t)BATCH * SEQ * DIMH);
}
